// PairwiseRankingLoss_50757923504321
// MI455X (gfx1250) — compile-verified, matrix operand order checked by reading, not run on hardware
//
#include <hip/hip_runtime.h>

// CDNA5 / gfx1250, wave32. WMMA vector types per probe-confirmed builtin signatures.
typedef __attribute__((ext_vector_type(16))) _Float16 v16h;
typedef __attribute__((ext_vector_type(8)))  _Float16 v8h;
typedef __attribute__((ext_vector_type(4)))  _Float16 v4h;
typedef __attribute__((ext_vector_type(8)))  float    v8f;

#define NROWS 1024     // N = 2 * 512 stacked views
#define DDIM  128
#define TEMP_INV 0.5f  // 1 / TEMPERATURE(=2)

// Row i of the virtual concatenated matrix f = [features[:,0]; features[:,1]]
// features memory layout: [512, 2, 128] row-major.
__device__ __forceinline__ const float* frow(const float* feat, int i) {
  int r = (i < 512) ? (2 * i) : (2 * (i - 512) + 1);
  return feat + r * DDIM;
}

// ---------------------------------------------------------------------------
// Fast path: precomputed hi/lo f16 matrices in d_ws (done once per row, not
// once per tile). ws layout: [0..4K) row norms f32, [4K..260K) hi f16,
// [260K..516K) lo f16, both row-major [1024][128].
// ---------------------------------------------------------------------------

// Kernel 1: per-row norms + fp32 -> (hi,lo) f16 split, and zero the output.
__global__ void pr_prep(const float* __restrict__ feat,
                        float* __restrict__ nrm,
                        _Float16* __restrict__ hi,
                        _Float16* __restrict__ lo,
                        float* __restrict__ out) {
  if (blockIdx.x == 0 && threadIdx.x == 0) out[0] = 0.0f;
  int wave = (blockIdx.x * blockDim.x + threadIdx.x) >> 5;  // one wave32 per row
  int lane = threadIdx.x & 31;
  if (wave >= NROWS) return;
  const float* row = frow(feat, wave);
  float4 a = ((const float4*)row)[lane];  // 32 lanes * 4 = 128 floats
  float v[4] = {a.x, a.y, a.z, a.w};
  v4h h, l;
  float s = 0.0f;
#pragma unroll
  for (int t = 0; t < 4; ++t) {
    s += v[t] * v[t];
    _Float16 hh = (_Float16)v[t];
    h[t] = hh;
    l[t] = (_Float16)(v[t] - (float)hh);  // residual for near-fp32 accuracy
  }
  *(v4h*)(hi + wave * DDIM + lane * 4) = h;
  *(v4h*)(lo + wave * DDIM + lane * 4) = l;
#pragma unroll
  for (int off = 16; off > 0; off >>= 1) s += __shfl_down(s, off, 32);
  if (lane == 0) nrm[wave] = s;
}

// Load 16 halves for one row in the CDNA5 16-bit A-matrix lane layout:
// lanes 0-15 hold row (lane&15) with K = kt+{0..7, 16..23},
// lanes 16-31 hold the same rows with K = kt+{8..15, 24..31}.
// Two contiguous 16-byte (b128) loads per call.
__device__ __forceinline__ v16h load16(const _Float16* __restrict__ base,
                                       int r, int kt, int koff) {
  v8h a = *(const v8h*)(base + r * DDIM + kt + koff);
  v8h b = *(const v8h*)(base + r * DDIM + kt + 16 + koff);
  v16h o;
#pragma unroll
  for (int t = 0; t < 8; ++t) { o[t] = a[t]; o[t + 8] = b[t]; }
  return o;
}

// Kernel 2: one wave per 16x16 Gram tile (upper triangle only), fused epilogue.
__global__ void pr_gram_loss(const _Float16* __restrict__ hi,
                             const _Float16* __restrict__ lo,
                             const float* __restrict__ labels,
                             const float* __restrict__ nrm,
                             float* __restrict__ out) {
  int wave = (blockIdx.x * blockDim.x + threadIdx.x) >> 5;
  int lane = threadIdx.x & 31;
  int ti = wave >> 6;   // row-tile index
  int tj = wave & 63;   // col-tile index
  if (tj < ti) return;  // wave-uniform: EXEC stays all-ones for WMMA waves

  int ra = ti * 16 + (lane & 15);
  int rb = tj * 16 + (lane & 15);    // B col n == f row n (Gram)
  int koff = (lane >> 4) << 3;       // 0 for lanes 0-15, 8 for lanes 16-31

  v8f acc = {};
#pragma unroll
  for (int kt = 0; kt < DDIM; kt += 32) {
    v16h Ah = load16(hi, ra, kt, koff);
    v16h Al = load16(lo, ra, kt, koff);
    v16h Bh = load16(hi, rb, kt, koff);
    v16h Bl = load16(lo, rb, kt, koff);
    // fp32-via-f16 split: G = Ah*Bh + Ah*Bl + Al*Bh  (Al*Bl ~ 2^-22, dropped)
    acc = __builtin_amdgcn_wmma_f32_16x16x32_f16(false, Ah, false, Bh,
                                                 (short)0, acc, false, false);
    acc = __builtin_amdgcn_wmma_f32_16x16x32_f16(false, Ah, false, Bl,
                                                 (short)0, acc, false, false);
    acc = __builtin_amdgcn_wmma_f32_16x16x32_f16(false, Al, false, Bh,
                                                 (short)0, acc, false, false);
  }

  // C/D layout: VGPR v, lane -> M = v + 8*(lane>>4), N = lane&15.
  int n = tj * 16 + (lane & 15);
  float nj = nrm[n];
  float lj = labels[n & 511];  // lab tiled: lab[i] = labels[i % 512]
  float sum = 0.0f;
#pragma unroll
  for (int v = 0; v < 8; ++v) {
    int m = ti * 16 + v + ((lane >> 4) << 3);
    if (n > m) {  // strict upper triangle (i < j)
      float d2 = nrm[m] + nj - 2.0f * acc[v];
      d2 = fmaxf(d2, 0.0f);  // guard fp rounding before sqrt
      float e = sqrtf(d2) * TEMP_INV - fabsf(labels[m & 511] - lj);
      sum += e * e;
    }
  }
#pragma unroll
  for (int off = 16; off > 0; off >>= 1) sum += __shfl_down(sum, off, 32);
  if (lane == 0) {
    // pre-scaled by 1/n_pairs, n_pairs = N(N-1)/2 = 523776
    atomicAdd(out, sum * (2.0f / (1024.0f * 1023.0f)));
  }
}

// ---------------------------------------------------------------------------
// Fallback path (tiny ws): fused conversion inside the gram kernel (round-1).
// ---------------------------------------------------------------------------

struct HiLo { v16h hi; v16h lo; };

__device__ __forceinline__ void cvt4(float4 a, _Float16* h, _Float16* l) {
  float v[4] = {a.x, a.y, a.z, a.w};
#pragma unroll
  for (int t = 0; t < 4; ++t) {
    _Float16 hh = (_Float16)v[t];
    h[t] = hh;
    l[t] = (_Float16)(v[t] - (float)hh);
  }
}

__device__ __forceinline__ HiLo load_tile_f32(const float* row, int kt, int koff) {
  float4 x0 = *(const float4*)(row + kt + koff);
  float4 x1 = *(const float4*)(row + kt + koff + 4);
  float4 x2 = *(const float4*)(row + kt + koff + 16);
  float4 x3 = *(const float4*)(row + kt + koff + 20);
  _Float16 h[16], l[16];
  cvt4(x0, h, l);
  cvt4(x1, h + 4, l + 4);
  cvt4(x2, h + 8, l + 8);
  cvt4(x3, h + 12, l + 12);
  HiLo r;
#pragma unroll
  for (int t = 0; t < 16; ++t) { r.hi[t] = h[t]; r.lo[t] = l[t]; }
  return r;
}

__global__ void pr_norms_init(const float* __restrict__ feat,
                              float* __restrict__ nrm,
                              float* __restrict__ out) {
  if (blockIdx.x == 0 && threadIdx.x == 0) out[0] = 0.0f;
  int wave = (blockIdx.x * blockDim.x + threadIdx.x) >> 5;
  int lane = threadIdx.x & 31;
  if (wave >= NROWS) return;
  const float* row = frow(feat, wave);
  float4 a = ((const float4*)row)[lane];
  float s = a.x * a.x + a.y * a.y + a.z * a.z + a.w * a.w;
#pragma unroll
  for (int off = 16; off > 0; off >>= 1) s += __shfl_down(s, off, 32);
  if (lane == 0) nrm[wave] = s;
}

__global__ void pr_gram_loss_fused(const float* __restrict__ feat,
                                   const float* __restrict__ labels,
                                   const float* __restrict__ nrm,
                                   float* __restrict__ out) {
  int wave = (blockIdx.x * blockDim.x + threadIdx.x) >> 5;
  int lane = threadIdx.x & 31;
  int ti = wave >> 6;
  int tj = wave & 63;
  if (tj < ti) return;

  const float* ra = frow(feat, ti * 16 + (lane & 15));
  const float* rb = frow(feat, tj * 16 + (lane & 15));
  int koff = (lane >> 4) << 3;

  v8f acc = {};
#pragma unroll
  for (int kt = 0; kt < DDIM; kt += 32) {
    HiLo A = load_tile_f32(ra, kt, koff);
    HiLo B = load_tile_f32(rb, kt, koff);
    acc = __builtin_amdgcn_wmma_f32_16x16x32_f16(false, A.hi, false, B.hi,
                                                 (short)0, acc, false, false);
    acc = __builtin_amdgcn_wmma_f32_16x16x32_f16(false, A.hi, false, B.lo,
                                                 (short)0, acc, false, false);
    acc = __builtin_amdgcn_wmma_f32_16x16x32_f16(false, A.lo, false, B.hi,
                                                 (short)0, acc, false, false);
  }

  int n = tj * 16 + (lane & 15);
  float nj = nrm[n];
  float lj = labels[n & 511];
  float sum = 0.0f;
#pragma unroll
  for (int v = 0; v < 8; ++v) {
    int m = ti * 16 + v + ((lane >> 4) << 3);
    if (n > m) {
      float d2 = nrm[m] + nj - 2.0f * acc[v];
      d2 = fmaxf(d2, 0.0f);
      float e = sqrtf(d2) * TEMP_INV - fabsf(labels[m & 511] - lj);
      sum += e * e;
    }
  }
#pragma unroll
  for (int off = 16; off > 0; off >>= 1) sum += __shfl_down(sum, off, 32);
  if (lane == 0) atomicAdd(out, sum * (2.0f / (1024.0f * 1023.0f)));
}

// ---------------------------------------------------------------------------

extern "C" void kernel_launch(void* const* d_in, const int* in_sizes, int n_in,
                              void* d_out, int out_size, void* d_ws, size_t ws_size,
                              hipStream_t stream) {
  const float* feat   = (const float*)d_in[0];  // [512, 2, 128] f32
  const float* labels = (const float*)d_in[1];  // [512, 1] f32
  float* out = (float*)d_out;                   // scalar f32

  const size_t NRM_BYTES = NROWS * sizeof(float);          // 4 KB
  const size_t MAT_BYTES = NROWS * DDIM * sizeof(_Float16); // 256 KB each
  const size_t NEED = NRM_BYTES + 2 * MAT_BYTES;           // 516 KB

  float* nrm = (float*)d_ws;

  if (ws_size >= NEED) {
    _Float16* hi = (_Float16*)((char*)d_ws + NRM_BYTES);
    _Float16* lo = (_Float16*)((char*)d_ws + NRM_BYTES + MAT_BYTES);
    // 1024 row waves -> 128 blocks * 8 waves; converts + norms + zeroes d_out.
    pr_prep<<<128, 256, 0, stream>>>(feat, nrm, hi, lo, out);
    // 64*64 = 4096 tile waves -> 512 blocks * 8 waves (lower-tri waves exit).
    pr_gram_loss<<<512, 256, 0, stream>>>(hi, lo, labels, nrm, out);
  } else {
    // Fallback: convert inside the gram kernel (needs only 4 KB of ws).
    pr_norms_init<<<128, 256, 0, stream>>>(feat, nrm, out);
    pr_gram_loss_fused<<<512, 256, 0, stream>>>(feat, labels, nrm, out);
  }
}
